// CommNetWork_40037685133905
// MI455X (gfx1250) — compile-verified, matrix operand order checked by reading, not run on hardware
//
#include <hip/hip_runtime.h>
#include <hip/hip_bf16.h>

typedef __attribute__((ext_vector_type(16))) __bf16 v16bf;
typedef __attribute__((ext_vector_type(8)))  __bf16 v8bf;
typedef __attribute__((ext_vector_type(8)))  float  v8f;

#define WMMA_BF16(a, b, c) \
    __builtin_amdgcn_wmma_f32_16x16x32_bf16(false, (a), false, (b), (short)0, (c), false, false)

// ---------------------------------------------------------------------------
// fp32 -> bf16 bulk convert (one-time pre-pass; n must be a multiple of 8)
// ---------------------------------------------------------------------------
__global__ __launch_bounds__(256)
void cvt_bf16_kernel(const float* __restrict__ src, __bf16* __restrict__ dst, int n) {
    const int i = (blockIdx.x * 256 + threadIdx.x) * 8;
    if (i >= n) return;
    float4 a = *reinterpret_cast<const float4*>(src + i);
    float4 b = *reinterpret_cast<const float4*>(src + i + 4);
    v8bf o;
    o[0] = (__bf16)a.x; o[1] = (__bf16)a.y; o[2] = (__bf16)a.z; o[3] = (__bf16)a.w;
    o[4] = (__bf16)b.x; o[5] = (__bf16)b.y; o[6] = (__bf16)b.z; o[7] = (__bf16)b.w;
    *reinterpret_cast<v8bf*>(dst + i) = o;
}

// ---------------------------------------------------------------------------
// 16x32 bf16 WMMA fragment from K-contiguous bf16 storage.
// ISA 7.12.2: lane L -> row = row0 + (L&15); K = (L>>4)*8 + {0..7, 16..23}.
// Two 16-byte (global_load_b128) loads per fragment, zero conversion VALU.
// ---------------------------------------------------------------------------
__device__ __forceinline__ v16bf load_frag(const __bf16* __restrict__ base,
                                           int ld, int row0, int k0, int lane) {
    const __bf16* p = base + (size_t)(row0 + (lane & 15)) * (size_t)ld
                           + (size_t)(k0 + ((lane >> 4) << 3));
    v8bf lo = *reinterpret_cast<const v8bf*>(p);
    v8bf hi = *reinterpret_cast<const v8bf*>(p + 16);
    v16bf f;
#pragma unroll
    for (int i = 0; i < 8; ++i) { f[i] = lo[i]; f[i + 8] = hi[i]; }
    return f;
}

// ---------------------------------------------------------------------------
// act(A[M,K]bf16 @ W[N,K]^T bf16 + bias[N]f32) -> optional f32 D / bf16 Db.
// Block: 256 thr = 8 waves, tile 128(M) x 64(N); wave -> 32x32 (2x2 tiles).
// ---------------------------------------------------------------------------
template <int RELU, int WF32, int WB16>
__global__ __launch_bounds__(256)
void gemm_bias_act_kernel(const __bf16* __restrict__ A, const __bf16* __restrict__ W,
                          const float* __restrict__ bias,
                          float* __restrict__ D, __bf16* __restrict__ Db,
                          int M, int N, int K) {
    const int lane = threadIdx.x & 31;
    const int wave = threadIdx.x >> 5;
    const int m0 = blockIdx.y * 128 + (wave & 3) * 32;
    const int n0 = blockIdx.x * 64  + (wave >> 2) * 32;

    v8f acc00 = {}, acc01 = {}, acc10 = {}, acc11 = {};
    for (int k0 = 0; k0 < K; k0 += 32) {
        v16bf a0 = load_frag(A, K, m0,      k0, lane);
        v16bf a1 = load_frag(A, K, m0 + 16, k0, lane);
        v16bf b0 = load_frag(W, K, n0,      k0, lane);
        v16bf b1 = load_frag(W, K, n0 + 16, k0, lane);
        acc00 = WMMA_BF16(a0, b0, acc00);
        acc01 = WMMA_BF16(a0, b1, acc01);
        acc10 = WMMA_BF16(a1, b0, acc10);
        acc11 = WMMA_BF16(a1, b1, acc11);
    }

    // C/D layout: VGPR r, lanes 0-15 -> M=r, lanes 16-31 -> M=8+r; N = lane&15.
    const int colA = n0 + (lane & 15);
    const int colB = colA + 16;
    const int rbase = (lane >> 4) << 3;
    const float bvA = bias[colA];
    const float bvB = bias[colB];
#pragma unroll
    for (int r = 0; r < 8; ++r) {
        const size_t rowT = (size_t)(m0 + rbase + r);
        const size_t rowB = rowT + 16;
        float v00 = acc00[r] + bvA, v01 = acc01[r] + bvB;
        float v10 = acc10[r] + bvA, v11 = acc11[r] + bvB;
        if (RELU) {
            v00 = v00 > 0.0f ? v00 : 0.0f;  v01 = v01 > 0.0f ? v01 : 0.0f;
            v10 = v10 > 0.0f ? v10 : 0.0f;  v11 = v11 > 0.0f ? v11 : 0.0f;
        }
        if (WF32) {
            D[rowT * N + colA] = v00;  D[rowT * N + colB] = v01;
            D[rowB * N + colA] = v10;  D[rowB * N + colB] = v11;
        }
        if (WB16) {
            Db[rowT * N + colA] = (__bf16)v00;  Db[rowT * N + colB] = (__bf16)v01;
            Db[rowB * N + colA] = (__bf16)v10;  Db[rowB * N + colB] = (__bf16)v11;
        }
    }
}

// ---------------------------------------------------------------------------
// c[b,n,:] = (sum_n' h[b,n',:] - h[b,n,:]) / 64  (fp32 in, bf16 out).
// ---------------------------------------------------------------------------
__global__ __launch_bounds__(256)
void comm_mean_kernel(const float* __restrict__ h, __bf16* __restrict__ c) {
    const int b = blockIdx.x;
    const int j = threadIdx.x;                       // H0 channel
    const float* hb = h + (size_t)b * 64 * 256;
    float s = 0.0f;
#pragma unroll 8
    for (int n = 0; n < 64; ++n) s += hb[(size_t)n * 256 + j];
    __bf16* cb = c + (size_t)b * 64 * 256;
#pragma unroll 8
    for (int n = 0; n < 64; ++n)
        cb[(size_t)n * 256 + j] = (__bf16)((s - hb[(size_t)n * 256 + j]) * (1.0f / 64.0f));
}

// ---------------------------------------------------------------------------
// Fused GRU: r/z accumulate both c@W_ih^T and h@W_hh^T into shared accums;
// n keeps gi_n and gh_n separate (needed for r*gh_n). Epilogue blends with
// fp32 h. Grid: (16 j-tiles, 128); 8 waves/block, one 16x16 tile per wave.
// ---------------------------------------------------------------------------
__global__ __launch_bounds__(256)
void gru_fused_kernel(const __bf16* __restrict__ c, const __bf16* __restrict__ hB,
                      const float* __restrict__ hF,
                      const __bf16* __restrict__ Wih, const float* __restrict__ bih,
                      const __bf16* __restrict__ Whh, const float* __restrict__ bhh,
                      __bf16* __restrict__ hnew) {
    const int lane = threadIdx.x & 31;
    const int wave = threadIdx.x >> 5;
    const int j0 = blockIdx.x * 16;
    const int m0 = (blockIdx.y * 8 + wave) * 16;

    v8f racc = {}, zacc = {}, niacc = {}, nhacc = {};
    for (int k0 = 0; k0 < 256; k0 += 32) {
        v16bf ac = load_frag(c,  256, m0, k0, lane);
        v16bf ah = load_frag(hB, 256, m0, k0, lane);
        v16bf b;
        b = load_frag(Wih, 256, j0,       k0, lane); racc  = WMMA_BF16(ac, b, racc);
        b = load_frag(Whh, 256, j0,       k0, lane); racc  = WMMA_BF16(ah, b, racc);
        b = load_frag(Wih, 256, j0 + 256, k0, lane); zacc  = WMMA_BF16(ac, b, zacc);
        b = load_frag(Whh, 256, j0 + 256, k0, lane); zacc  = WMMA_BF16(ah, b, zacc);
        b = load_frag(Wih, 256, j0 + 512, k0, lane); niacc = WMMA_BF16(ac, b, niacc);
        b = load_frag(Whh, 256, j0 + 512, k0, lane); nhacc = WMMA_BF16(ah, b, nhacc);
    }

    const int col = j0 + (lane & 15);
    const int rbase = (lane >> 4) << 3;
    const float br  = bih[col]       + bhh[col];
    const float bz  = bih[col + 256] + bhh[col + 256];
    const float bin = bih[col + 512];
    const float bhn = bhh[col + 512];
#pragma unroll
    for (int r = 0; r < 8; ++r) {
        const size_t row = (size_t)(m0 + rbase + r);
        const float rr = 1.0f / (1.0f + __expf(-(racc[r] + br)));
        const float zz = 1.0f / (1.0f + __expf(-(zacc[r] + bz)));
        const float nn = tanhf((niacc[r] + bin) + rr * (nhacc[r] + bhn));
        const float ho = hF[row * 256 + col];
        hnew[row * 256 + col] = (__bf16)((1.0f - zz) * nn + zz * ho);
    }
}

// ---------------------------------------------------------------------------
extern "C" void kernel_launch(void* const* d_in, const int* in_sizes, int n_in,
                              void* d_out, int out_size, void* d_ws, size_t ws_size,
                              hipStream_t stream) {
    const float* obs   = (const float*)d_in[0];
    const float* W_enc = (const float*)d_in[1];
    const float* b_enc = (const float*)d_in[2];
    const float* W_obs = (const float*)d_in[3];
    const float* b_obs = (const float*)d_in[4];
    const float* W_ih  = (const float*)d_in[5];
    const float* b_ih  = (const float*)d_in[6];
    const float* W_hh  = (const float*)d_in[7];
    const float* b_hh  = (const float*)d_in[8];
    const float* W_val = (const float*)d_in[9];
    const float* b_val = (const float*)d_in[10];
    const float* W_dec = (const float*)d_in[11];
    const float* b_dec = (const float*)d_in[12];
    float* out = (float*)d_out;

    const int M = 256 * 64;                       // B*N = 16384 rows
    char* w = (char*)d_ws;
    size_t off = 0;
    auto carve = [&](size_t bytes) { char* p = w + off; off += (bytes + 255) & ~(size_t)255; return p; };

    __bf16* wb_enc = (__bf16*)carve((size_t)256 * 128 * 2);
    __bf16* wb_obs = (__bf16*)carve((size_t)256 * 256 * 2);
    __bf16* wb_ih  = (__bf16*)carve((size_t)768 * 256 * 2);
    __bf16* wb_hh  = (__bf16*)carve((size_t)768 * 256 * 2);
    __bf16* wb_val = (__bf16*)carve((size_t)256 * 256 * 2);
    __bf16* wb_dec = (__bf16*)carve((size_t)64  * 256 * 2);
    __bf16* obs_b  = (__bf16*)carve((size_t)M * 128 * 2);
    __bf16* slot1  = (__bf16*)carve((size_t)M * 256 * 2);   // x_b -> c_b -> v_b
    __bf16* h_b    = (__bf16*)carve((size_t)M * 256 * 2);
    float*  h_f    = (float*) carve((size_t)M * 256 * 4);
    __bf16* hn_b   = (__bf16*)carve((size_t)M * 256 * 2);

    dim3 blk(256);
    auto cvt = [&](const float* s, __bf16* d, int n) {
        cvt_bf16_kernel<<<dim3((n / 8 + 255) / 256), blk, 0, stream>>>(s, d, n);
    };
    // one-time bf16 conversion of obs + all weights
    cvt(obs,   obs_b,  M * 128);
    cvt(W_enc, wb_enc, 256 * 128);
    cvt(W_obs, wb_obs, 256 * 256);
    cvt(W_ih,  wb_ih,  768 * 256);
    cvt(W_hh,  wb_hh,  768 * 256);
    cvt(W_val, wb_val, 256 * 256);
    cvt(W_dec, wb_dec, 64 * 256);

    // x = relu(obs @ W_enc^T + b_enc)           -> bf16 only
    gemm_bias_act_kernel<1, 0, 1><<<dim3(4, M / 128), blk, 0, stream>>>(
        obs_b, wb_enc, b_enc, nullptr, slot1, M, 256, 128);
    // h = x @ W_obs^T + b_obs                   -> fp32 (comm/blend) + bf16 (frags)
    gemm_bias_act_kernel<0, 1, 1><<<dim3(4, M / 128), blk, 0, stream>>>(
        slot1, wb_obs, b_obs, h_f, h_b, M, 256, 256);
    // c = (sum_agents - h)/N                    (x dead -> reuse slot1)
    comm_mean_kernel<<<dim3(256), blk, 0, stream>>>(h_f, slot1);
    // fused GRU -> h_new (bf16)
    gru_fused_kernel<<<dim3(16, M / 128), blk, 0, stream>>>(
        slot1, h_b, h_f, wb_ih, b_ih, wb_hh, b_hh, hn_b);
    // v = h_new @ W_val^T + b_val               (c dead -> reuse slot1)
    gemm_bias_act_kernel<0, 0, 1><<<dim3(4, M / 128), blk, 0, stream>>>(
        hn_b, wb_val, b_val, nullptr, slot1, M, 256, 256);
    // out = v @ W_dec^T + b_dec                 -> fp32 d_out
    gemm_bias_act_kernel<0, 1, 0><<<dim3(1, M / 128), blk, 0, stream>>>(
        slot1, wb_dec, b_dec, out, nullptr, M, 64, 256);
}
